// SMamba_67817533604175
// MI455X (gfx1250) — matrix operand order repaired and checked by reading, not compiled, not run on hardware
//
#include <hip/hip_runtime.h>
#include <hip/hip_bf16.h>
#include <math.h>

// ---------------------------------------------------------------------------
// S-Mamba forward on gfx1250: bf16 WMMA GEMMs (pre-converted bf16 operands)
// + register-resident selective scan.
// ---------------------------------------------------------------------------

#define BATCH 32
#define SEQL  64
#define LTOK  1024        // NVAR -> token/sequence length inside mamba
#define HDIM  512
#define DSTATE 32
#define DTRANK 32
#define DFFN  128
#define XDW   (DTRANK + 2 * DSTATE)    // 96
#define NROWS (BATCH * LTOK)           // 32768

typedef __attribute__((ext_vector_type(16))) __bf16 v16bf;
typedef __attribute__((ext_vector_type(8)))  __bf16 v8bf;
typedef __attribute__((ext_vector_type(8)))  float  v8f;

static __device__ inline __bf16 f2bf(float f) {
    unsigned u = __builtin_bit_cast(unsigned, f);
    unsigned r = u + 0x7FFFu + ((u >> 16) & 1u);   // round-to-nearest-even
    unsigned short h = (unsigned short)(r >> 16);
    return __builtin_bit_cast(__bf16, h);
}

static __device__ inline unsigned pack2bf(float lo, float hi) {
    unsigned a = (unsigned)__builtin_bit_cast(unsigned short, f2bf(lo));
    unsigned b = (unsigned)__builtin_bit_cast(unsigned short, f2bf(hi));
    return a | (b << 16);
}

// ---------------------------------------------------------------------------
// f32 -> bf16 bulk conversion (8 elements / thread, 16B stores)
// ---------------------------------------------------------------------------
__global__ __launch_bounds__(256) void cvt_bf16_kernel(
    const float* __restrict__ src, __bf16* __restrict__ dst, int n8)
{
    int i = blockIdx.x * 256 + threadIdx.x;
    if (i >= n8) return;
    const float4* s = reinterpret_cast<const float4*>(src) + (size_t)i * 2;
    float4 a = s[0], b = s[1];
    uint4 o;
    o.x = pack2bf(a.x, a.y);
    o.y = pack2bf(a.z, a.w);
    o.z = pack2bf(b.x, b.y);
    o.w = pack2bf(b.z, b.w);
    reinterpret_cast<uint4*>(dst)[i] = o;
}

// ---------------------------------------------------------------------------
// Tiled WMMA GEMM on bf16 operands (no in-loop guards; exact tiling):
//   val = sum_k A[m,k]*W[n,k] + bias[n];  act;  Out(f32) and/or OutBf(bf16)
// A: [M, lda] bf16, W: [N, K] bf16 row-major.
// flip!=0: A-row m=(b,l) reads source row (b, Lf-1-l)  (folds jnp.flip).
// Block = 256 = 8 waves; wave -> 16 rows x (16*G) cols, A fragment reused xG.
// ACT: 0=none, 1=exact GELU, 2=softplus.  G in {2,4}.
// Requires M % 128 == 0, N % (16*G) == 0, K % 32 == 0.
// ---------------------------------------------------------------------------
template <int ACT, int G>
__global__ __launch_bounds__(256) void gemm_wmma_kernel(
    const __bf16* __restrict__ A, int lda,
    const __bf16* __restrict__ W,
    const float* __restrict__ bias,
    float* __restrict__ Out, __bf16* __restrict__ OutBf, int ldo,
    int M, int N, int K, int flip, int Lf)
{
    const int wave = threadIdx.x >> 5;
    const int lane = threadIdx.x & 31;
    const int half = lane >> 4;
    const int lrow = lane & 15;

    const int rowBase = blockIdx.x * 128 + wave * 16;
    const int colBase = blockIdx.y * (16 * G);

    int m = rowBase + lrow;
    int srcRow = m;
    if (flip) {
        int bb = m / Lf;
        int ll = m - bb * Lf;
        srcRow = bb * Lf + (Lf - 1 - ll);
    }
    // A fragment layout (16-bit A 16x32): lane half selects k-base {0,8};
    // vector elems 0..7 -> k = ca+0..7, elems 8..15 -> k = ca+16..23.
    const v8bf* Arow8 = reinterpret_cast<const v8bf*>(A + (size_t)srcRow * lda);
    const int ca8 = half;                 // (ca = 8*half)/8
    const int cb  = half ? 8 : 0;         // B: k = cb + {0..7}, cb + {16..23}

    v8f acc[G];
#pragma unroll
    for (int g = 0; g < G; ++g)
#pragma unroll
        for (int e = 0; e < 8; ++e) acc[g][e] = 0.0f;

    const __bf16* Wbase = W + (size_t)(colBase + lrow) * K + cb;

    for (int k0 = 0; k0 < K; k0 += 32) {
        int b8 = (k0 >> 3) + ca8;
        v8bf alo = Arow8[b8];
        v8bf ahi = Arow8[b8 + 2];
        v16bf af = __builtin_shufflevector(alo, ahi,
            0, 1, 2, 3, 4, 5, 6, 7, 8, 9, 10, 11, 12, 13, 14, 15);

#pragma unroll
        for (int g = 0; g < G; ++g) {
            const v8bf* Wr = reinterpret_cast<const v8bf*>(
                Wbase + (size_t)g * 16 * K + k0);
            v8bf blo = Wr[0];
            v8bf bhi = Wr[2];
            v16bf bfv = __builtin_shufflevector(blo, bhi,
                0, 1, 2, 3, 4, 5, 6, 7, 8, 9, 10, 11, 12, 13, 14, 15);
            acc[g] = __builtin_amdgcn_wmma_f32_16x16x32_bf16(
                false, af, false, bfv, (short)0, acc[g], false, false);
        }
    }

#pragma unroll
    for (int g = 0; g < G; ++g) {
        int nc = colBase + g * 16 + lrow;
        float bv = bias ? bias[nc] : 0.0f;
#pragma unroll
        for (int v = 0; v < 8; ++v) {
            int row = rowBase + v + (half ? 8 : 0);
            float val = acc[g][v] + bv;
            if (ACT == 1) {
                val = 0.5f * val * (1.0f + erff(val * 0.70710678118654752f));
            } else if (ACT == 2) {
                val = (val > 20.0f) ? val : log1pf(__expf(val));
            }
            size_t o = (size_t)row * ldo + nc;
            if (Out)   Out[o]   = val;
            if (OutBf) OutBf[o] = f2bf(val);
        }
    }
}

// --------------------------- elementwise kernels ---------------------------

// x_enc[b][s][n] (n fastest) -> At_bf16[(b*LTOK + n)*SEQL + s]
__global__ __launch_bounds__(256) void transpose_embed_kernel(
    const float* __restrict__ x, __bf16* __restrict__ At, int total)
{
    int i = blockIdx.x * 256 + threadIdx.x;
    if (i >= total) return;
    int n = i & (LTOK - 1);
    int s = (i >> 10) & (SEQL - 1);
    int b = i >> 16;
    At[((size_t)(b * LTOK + n)) * SEQL + s] = f2bf(x[i]);
}

// depthwise conv(k=2) + bias + SiLU on x-half of xz -> xconv (f32 + bf16)
__global__ __launch_bounds__(256) void conv_silu_kernel(
    const float* __restrict__ xz, const float* __restrict__ cw,
    const float* __restrict__ cbias, float* __restrict__ xconv,
    __bf16* __restrict__ xconv_bf, int total)
{
    int i = blockIdx.x * 256 + threadIdx.x;
    if (i >= total) return;
    int d  = i & (HDIM - 1);
    int rl = i >> 9;
    int l  = rl & (LTOK - 1);
    size_t base = (size_t)rl * (2 * HDIM);
    float xc = xz[base + d];
    float xp = (l > 0) ? xz[base - 2 * HDIM + d] : 0.0f;
    float v = xp * cw[2 * d] + xc * cw[2 * d + 1] + cbias[d];
    float r = v / (1.0f + __expf(-v));
    xconv[i]    = r;
    xconv_bf[i] = f2bf(r);
}

// Selective scan: grid=BATCH, block=HDIM; h[32] in registers, B/C via LDS.
// Fuses y = (scan + x*D) * silu(z); writes bf16 (consumed only by out_proj).
__global__ __launch_bounds__(512) void scan_kernel(
    const float* __restrict__ xz, const float* __restrict__ xconv,
    const float* __restrict__ xdbl, const float* __restrict__ dt_in,
    __bf16* __restrict__ ybf,
    const float* __restrict__ A_log, const float* __restrict__ Dv)
{
    const int b = blockIdx.x;
    const int d = threadIdx.x;

    float a[DSTATE], h[DSTATE];
#pragma unroll
    for (int s = 0; s < DSTATE; ++s) {
        a[s] = -__expf(A_log[d * DSTATE + s]);
        h[s] = 0.0f;
    }
    const float dvec = Dv[d];

    __shared__ float Bs[DSTATE], Cs[DSTATE];

    for (int l = 0; l < LTOK; ++l) {
        int rl = (b << 10) + l;
        __syncthreads();
        const float* xd = xdbl + (size_t)rl * XDW;
        if (d < 64) {
            if (d < 32) Bs[d] = xd[DTRANK + d];
            else        Cs[d - 32] = xd[DTRANK + d];
        }
        __syncthreads();

        size_t idx = ((size_t)rl << 9) + d;
        float dt = dt_in[idx];
        float xv = xconv[idx];
        float z  = xz[((size_t)rl << 10) + HDIM + d];
        float dtx = dt * xv;
        float acc = 0.0f;
#pragma unroll
        for (int s = 0; s < DSTATE; ++s) {
            h[s] = h[s] * __expf(dt * a[s]) + dtx * Bs[s];
            acc += h[s] * Cs[s];
        }
        float sz = z / (1.0f + __expf(-z));
        ybf[idx] = f2bf((acc + xv * dvec) * sz);
    }
}

// enc += tmpF + flip_L(tmpB)
__global__ __launch_bounds__(256) void resadd_kernel(
    float* __restrict__ enc, const float* __restrict__ tf,
    const float* __restrict__ tb, int total)
{
    int i = blockIdx.x * 256 + threadIdx.x;
    if (i >= total) return;
    int d  = i & (HDIM - 1);
    int rl = i >> 9;
    int b  = rl >> 10;
    int l  = rl & (LTOK - 1);
    int rlB = (b << 10) + (LTOK - 1 - l);
    enc[i] += tf[i] + tb[((size_t)rlB << 9) + d];
}

// In-place LayerNorm over H=512, optional fused residual, optional bf16 shadow
__global__ __launch_bounds__(256) void ln_kernel(
    float* __restrict__ x, const float* __restrict__ res,
    const float* __restrict__ g, const float* __restrict__ bt,
    __bf16* __restrict__ xb)
{
    int row = blockIdx.x;
    int t = threadIdx.x;
    float* xr = x + (size_t)row * HDIM;
    float v0 = xr[t];
    float v1 = xr[t + 256];
    if (res) {
        v0 += res[(size_t)row * HDIM + t];
        v1 += res[(size_t)row * HDIM + t + 256];
    }
    __shared__ float s1[256], s2[256];
    s1[t] = v0 + v1;
    s2[t] = v0 * v0 + v1 * v1;
    __syncthreads();
    for (int off = 128; off > 0; off >>= 1) {
        if (t < off) { s1[t] += s1[t + off]; s2[t] += s2[t + off]; }
        __syncthreads();
    }
    float mean = s1[0] * (1.0f / HDIM);
    float var  = s2[0] * (1.0f / HDIM) - mean * mean;
    float rstd = rsqrtf(var + 1e-5f);
    float o0 = (v0 - mean) * rstd * g[t]       + bt[t];
    float o1 = (v1 - mean) * rstd * g[t + 256] + bt[t + 256];
    xr[t]       = o0;
    xr[t + 256] = o1;
    if (xb) {
        xb[(size_t)row * HDIM + t]       = f2bf(o0);
        xb[(size_t)row * HDIM + t + 256] = f2bf(o1);
    }
}

// out[row] = dot(x[row,:], w) + b0   (one wave per row, shuffle reduce)
__global__ __launch_bounds__(256) void proj_kernel(
    const float* __restrict__ x, const float* __restrict__ w,
    const float* __restrict__ b, float* __restrict__ out, int rows)
{
    int row  = blockIdx.x * 8 + (threadIdx.x >> 5);
    int lane = threadIdx.x & 31;
    if (row >= rows) return;
    const float* xr = x + (size_t)row * HDIM;
    float s = 0.0f;
#pragma unroll
    for (int i = lane; i < HDIM; i += 32) s += xr[i] * w[i];
#pragma unroll
    for (int off = 16; off > 0; off >>= 1) s += __shfl_down(s, off, 32);
    if (lane == 0) out[row] = s + b[0];
}

// ------------------------------- host side ---------------------------------

static void gemm(int act, const __bf16* A, int lda, const __bf16* W,
                 const float* bias, float* Out, __bf16* OutBf, int ldo,
                 int M, int N, int K, int flip, int Lf, hipStream_t s)
{
    dim3 block(256);
    if (N % 64 == 0) {
        dim3 grid(M / 128, N / 64);
        if (act == 0)
            gemm_wmma_kernel<0, 4><<<grid, block, 0, s>>>(A, lda, W, bias, Out, OutBf, ldo, M, N, K, flip, Lf);
        else if (act == 1)
            gemm_wmma_kernel<1, 4><<<grid, block, 0, s>>>(A, lda, W, bias, Out, OutBf, ldo, M, N, K, flip, Lf);
        else
            gemm_wmma_kernel<2, 4><<<grid, block, 0, s>>>(A, lda, W, bias, Out, OutBf, ldo, M, N, K, flip, Lf);
    } else {   // N % 32 == 0  (x_proj: N=96)
        dim3 grid(M / 128, N / 32);
        if (act == 0)
            gemm_wmma_kernel<0, 2><<<grid, block, 0, s>>>(A, lda, W, bias, Out, OutBf, ldo, M, N, K, flip, Lf);
        else if (act == 1)
            gemm_wmma_kernel<1, 2><<<grid, block, 0, s>>>(A, lda, W, bias, Out, OutBf, ldo, M, N, K, flip, Lf);
        else
            gemm_wmma_kernel<2, 2><<<grid, block, 0, s>>>(A, lda, W, bias, Out, OutBf, ldo, M, N, K, flip, Lf);
    }
}

extern "C" void kernel_launch(void* const* d_in, const int* in_sizes, int n_in,
                              void* d_out, int out_size, void* d_ws, size_t ws_size,
                              hipStream_t stream)
{
    (void)in_sizes; (void)n_in; (void)out_size; (void)ws_size;
    const int R = NROWS;
    auto P = [&](int i) { return (const float*)d_in[i]; };

    // ---- workspace layout: f32 region, then 16B-aligned bf16 region ----
    float* fbase = (float*)d_ws;
    size_t foff = 0;
    auto allocF = [&](size_t n) { float* p = fbase + foff; foff += n; return p; };

    float* ENC   = allocF((size_t)R * HDIM);
    float* TMPF  = allocF((size_t)R * HDIM);
    float* TMPB  = allocF((size_t)R * HDIM);
    float* XZ    = allocF((size_t)R * 2 * HDIM);
    float* XCONV = allocF((size_t)R * HDIM);
    float* XDBL  = allocF((size_t)R * XDW);
    float* DTY   = allocF((size_t)R * HDIM);

    __bf16* bbase = (__bf16*)(fbase + foff);
    size_t boff = 0;
    auto allocB = [&](size_t n) { __bf16* p = bbase + boff; boff += n; return p; };

    __bf16* AT_BF    = allocB((size_t)R * SEQL);
    __bf16* ENC_BF   = allocB((size_t)R * HDIM);
    __bf16* XCONV_BF = allocB((size_t)R * HDIM);
    __bf16* XDBL_BF  = allocB((size_t)R * XDW);
    __bf16* Y_BF     = allocB((size_t)R * HDIM);
    __bf16* FF1_BF   = allocB((size_t)R * DFFN);

    // ---- convert all GEMM weights to bf16 once ----
    auto cvtW = [&](int idx, size_t n) {
        __bf16* dst = allocB(n);
        int n8 = (int)(n / 8);
        cvt_bf16_kernel<<<(n8 + 255) / 256, 256, 0, stream>>>(P(idx), dst, n8);
        return dst;
    };

    __bf16* W_EMB = cvtW(1, (size_t)HDIM * SEQL);
    __bf16 *W_IN[2][2], *W_XP[2][2], *W_DT[2][2], *W_OP[2][2];
    __bf16 *W_FF1[2], *W_FF2[2];
    for (int layer = 0; layer < 2; ++layer) {
        int base = 5 + layer * 26;
        for (int dir = 0; dir < 2; ++dir) {
            int mb = base + dir * 9;
            W_IN[layer][dir] = cvtW(mb + 0, (size_t)2 * HDIM * HDIM);
            W_XP[layer][dir] = cvtW(mb + 3, (size_t)XDW * HDIM);
            W_DT[layer][dir] = cvtW(mb + 4, (size_t)HDIM * DTRANK);
            W_OP[layer][dir] = cvtW(mb + 8, (size_t)HDIM * HDIM);
        }
        W_FF1[layer] = cvtW(base + 22, (size_t)DFFN * HDIM);
        W_FF2[layer] = cvtW(base + 24, (size_t)HDIM * DFFN);
    }

    // ---- embedding: enc = transpose(x_enc) @ emb_w.T + emb_b ----
    {
        int total = BATCH * SEQL * LTOK;
        transpose_embed_kernel<<<(total + 255) / 256, 256, 0, stream>>>(P(0), AT_BF, total);
    }
    gemm(0, AT_BF, SEQL, W_EMB, P(2), ENC, nullptr, HDIM, R, HDIM, SEQL, 0, LTOK, stream);
    ln_kernel<<<R, 256, 0, stream>>>(ENC, nullptr, P(3), P(4), ENC_BF);   // norm0

    // ---- encoder layers ----
    for (int layer = 0; layer < 2; ++layer) {
        int base = 5 + layer * 26;
        for (int dir = 0; dir < 2; ++dir) {
            int mb = base + dir * 9;
            float* OUTBUF = dir ? TMPB : TMPF;
            // xz = u @ in_proj.T   (flip folded into A-row indexing for dir=1)
            gemm(0, ENC_BF, HDIM, W_IN[layer][dir], nullptr, XZ, nullptr, 2 * HDIM,
                 R, 2 * HDIM, HDIM, dir, LTOK, stream);
            conv_silu_kernel<<<(R * HDIM + 255) / 256, 256, 0, stream>>>(
                XZ, P(mb + 1), P(mb + 2), XCONV, XCONV_BF, R * HDIM);
            // xdbl = x @ x_proj.T   (f32 for scan B/C + bf16 for dt GEMM)
            gemm(0, XCONV_BF, HDIM, W_XP[layer][dir], nullptr, XDBL, XDBL_BF, XDW,
                 R, XDW, HDIM, 0, LTOK, stream);
            // dt = softplus(xdbl[:, :32] @ dt_w.T + dt_b)
            gemm(2, XDBL_BF, XDW, W_DT[layer][dir], P(mb + 5), DTY, nullptr, HDIM,
                 R, HDIM, DTRANK, 0, LTOK, stream);
            // selective scan; fuses +x*D and *silu(z); emits bf16 y
            scan_kernel<<<BATCH, HDIM, 0, stream>>>(
                XZ, XCONV, XDBL, DTY, Y_BF, P(mb + 6), P(mb + 7));
            // out = y @ out_proj.T
            gemm(0, Y_BF, HDIM, W_OP[layer][dir], nullptr, OUTBUF, nullptr, HDIM,
                 R, HDIM, HDIM, 0, LTOK, stream);
        }
        resadd_kernel<<<(R * HDIM + 255) / 256, 256, 0, stream>>>(ENC, TMPF, TMPB, R * HDIM);
        ln_kernel<<<R, 256, 0, stream>>>(ENC, nullptr, P(base + 18), P(base + 19), ENC_BF); // n1
        // FFN
        gemm(1, ENC_BF, HDIM, W_FF1[layer], P(base + 23), nullptr, FF1_BF, DFFN,
             R, DFFN, HDIM, 0, LTOK, stream);                                   // ff1 + GELU (bf16 only)
        gemm(0, FF1_BF, DFFN, W_FF2[layer], P(base + 25), TMPF, nullptr, HDIM,
             R, HDIM, DFFN, 0, LTOK, stream);                                   // ff2
        ln_kernel<<<R, 256, 0, stream>>>(ENC, TMPF, P(base + 20), P(base + 21), ENC_BF); // n2(x + ff)
    }

    // ---- final norm + projection to 1 ----
    ln_kernel<<<R, 256, 0, stream>>>(ENC, nullptr, P(57), P(58), nullptr);
    proj_kernel<<<R / 8, 256, 0, stream>>>(ENC, P(59), P(60), (float*)d_out, R);
}
